// MoE_19679540150991
// MI455X (gfx1250) — compile-verified, matrix operand order checked by reading, not run on hardware
//
#include <hip/hip_runtime.h>

// ---------------------------------------------------------------------------
// MoE top-2 SwiGLU for MI455X (gfx1250, wave32, WMMA bf16 + async LDS loads)
// B=4 S=2048 C=1024 E=8 F=2752 K=2  -> T=8192 tokens
// M-tile = 64 tokens: each weight B-fragment feeds FOUR WMMAs (4 row-blocks),
// quartering per-FLOP L2 weight traffic, the binding resource once the 135 MB
// of bf16 weights are resident in the 192 MB L2.
// LDS: ~224 KB of the WGP's 320 KB (large-LDS confirmed to compile on gfx1250).
// Workspace requirement: ~153 MB (x_bf16 16MB, wgT/wuT/wdT 135MB, routing ~0.5MB)
// ---------------------------------------------------------------------------

#define DEVFN __device__ __forceinline__

constexpr int kB = 4, kS = 2048, kC = 1024, kE = 8, kF = 2752;
constexpr int kT = kB * kS;             // 8192 tokens
constexpr int kMTile = 64;              // tokens per workgroup tile
constexpr int kMBlk  = kMTile / 16;     // 4 row-blocks of 16
constexpr int kMaxTiles = kT / kMTile;  // 128 row tiles per expert (worst case)
constexpr int kFChunk = 704;            // F processed in chunks (704,704,704,640)

typedef __bf16 bf16;
typedef __attribute__((ext_vector_type(16))) __bf16 bf16x16;
typedef __attribute__((ext_vector_type(8)))  __bf16 bf16x8;
typedef __attribute__((ext_vector_type(8)))  float  v8f;

// D = A(16x32 bf16) * B(32x16 bf16) + C(16x16 f32)
DEVFN v8f wmma_bf16(bf16x16 a, bf16x16 b, v8f c) {
    return __builtin_amdgcn_wmma_f32_16x16x32_bf16(
        /*neg_a=*/false, a, /*neg_b=*/false, b,
        /*c_mod=*/(short)0, c, /*reuse_a=*/false, /*reuse_b=*/false);
}

// Load one 16x32 (A) or 32x16 (B) bf16 fragment.
// Per-lane layout (ISA 7.12.2): lane L holds row/col (L&15), K-base (L>>4)*8,
// as two contiguous runs of 8 bf16: [kb, kb+8) and [kb+16, kb+24).
DEVFN bf16x16 load_frag(const bf16* base, int stride, int lane) {
    const int n  = lane & 15;
    const int kb = (lane >> 4) << 3;
    const bf16* p = base + (size_t)n * stride + kb;
    bf16x8 lo = *reinterpret_cast<const bf16x8*>(p);
    bf16x8 hi = *reinterpret_cast<const bf16x8*>(p + 16);
    return __builtin_shufflevector(lo, hi, 0, 1, 2, 3, 4, 5, 6, 7,
                                           8, 9, 10, 11, 12, 13, 14, 15);
}

// Async global->LDS 16-byte copy (GLOBAL_LOAD_ASYNC_TO_LDS_B128, ASYNCcnt).
// lds_off: byte offset within the workgroup LDS allocation (generic ptr low bits).
DEVFN void async_copy_b128(unsigned lds_off, const void* gaddr) {
    asm volatile("global_load_async_to_lds_b128 %0, %1, off"
                 :: "v"(lds_off), "v"(gaddr) : "memory");
}
DEVFN void wait_asynccnt0() {
    asm volatile("s_wait_asynccnt 0x0" ::: "memory");
}

DEVFN float fast_silu_mul(float g, float u) {
    // silu(g)*u with v_rcp_f32 instead of the IEEE division expansion
    return g * u * __builtin_amdgcn_rcpf(1.0f + __expf(-g));
}

// ---------------------------------------------------------------------------
// Kernel 0: zero output + routing counters (graph-safe, re-runs every call)
// ---------------------------------------------------------------------------
__global__ void moe_zero_kernel(float* __restrict__ out, int* __restrict__ counts,
                                size_t n) {
    size_t i = (size_t)blockIdx.x * blockDim.x + threadIdx.x;
    const size_t stride = (size_t)gridDim.x * blockDim.x;
    for (; i < n; i += stride) out[i] = 0.0f;
    if (blockIdx.x == 0 && threadIdx.x < 64) counts[threadIdx.x] = 0;
}

// ---------------------------------------------------------------------------
// Kernel 1: fp32 -> bf16 elementwise (activations)
// ---------------------------------------------------------------------------
__global__ void moe_cvt_bf16_kernel(const float* __restrict__ in,
                                    bf16* __restrict__ out, size_t n) {
    size_t i = (size_t)blockIdx.x * blockDim.x + threadIdx.x;
    const size_t stride = (size_t)gridDim.x * blockDim.x;
    for (; i < n; i += stride) out[i] = (bf16)in[i];
}

// ---------------------------------------------------------------------------
// Kernel 2: fp32 [E][rows][cols] -> bf16 [E][cols][rows] (transpose so the
// GEMM B-fragments are contiguous along K for 16-byte loads)
// ---------------------------------------------------------------------------
__global__ void moe_transpose_bf16_kernel(const float* __restrict__ in,
                                          bf16* __restrict__ out,
                                          int rows, int cols) {
    const size_t plane = (size_t)rows * cols;
    const size_t n = plane * kE;
    size_t i = (size_t)blockIdx.x * blockDim.x + threadIdx.x;
    const size_t stride = (size_t)gridDim.x * blockDim.x;
    for (; i < n; i += stride) {
        const size_t e = i / plane;
        const size_t rem = i - e * plane;
        const int r = (int)(rem / cols);
        const int c = (int)(rem - (size_t)r * cols);
        out[e * plane + (size_t)c * rows + r] = (bf16)in[i];
    }
}

// ---------------------------------------------------------------------------
// Kernel 3: gating. One wave32 per token: logits over E=8, softmax, top-2,
// normalized weights, atomic append to per-expert token lists.
// ---------------------------------------------------------------------------
__global__ __launch_bounds__(256)
void moe_gate_kernel(const float* __restrict__ x,
                     const float* __restrict__ wgate,
                     int* __restrict__ counts,
                     int* __restrict__ tok_idx,
                     float* __restrict__ tok_w) {
    const int lane  = threadIdx.x & 31;
    const int wv    = threadIdx.x >> 5;
    const int token = blockIdx.x * 8 + wv;
    if (token >= kT) return;   // wave-uniform

    float s[kE];
#pragma unroll
    for (int e = 0; e < kE; ++e) s[e] = 0.0f;

    for (int c = lane; c < kC; c += 32) {
        const float xv = x[(size_t)token * kC + c];
#pragma unroll
        for (int e = 0; e < kE; ++e) s[e] += xv * wgate[e * kC + c];
    }
    // wave32 cross-lane reduction
#pragma unroll
    for (int e = 0; e < kE; ++e) {
#pragma unroll
        for (int off = 16; off > 0; off >>= 1) s[e] += __shfl_xor(s[e], off, 32);
    }
    // softmax over 8 experts
    float m = s[0];
#pragma unroll
    for (int e = 1; e < kE; ++e) m = fmaxf(m, s[e]);
    float p[kE], d = 0.0f;
#pragma unroll
    for (int e = 0; e < kE; ++e) { p[e] = __expf(s[e] - m); d += p[e]; }
    const float inv = __builtin_amdgcn_rcpf(d);
#pragma unroll
    for (int e = 0; e < kE; ++e) p[e] *= inv;

    // top-2
    int i0 = 0; float s0 = p[0];
#pragma unroll
    for (int e = 1; e < kE; ++e) if (p[e] > s0) { s0 = p[e]; i0 = e; }
    int i1 = (i0 == 0) ? 1 : 0; float s1 = p[i1];
#pragma unroll
    for (int e = 0; e < kE; ++e) if (e != i0 && p[e] > s1) { s1 = p[e]; i1 = e; }

    const float wnorm = __builtin_amdgcn_rcpf(s0 + s1 + 1e-9f);
    const float w0 = s0 * wnorm, w1 = s1 * wnorm;

    if (lane == 0) {
        int slot0 = atomicAdd(&counts[i0], 1);
        tok_idx[i0 * kT + slot0] = token;
        tok_w  [i0 * kT + slot0] = w0;
        int slot1 = atomicAdd(&counts[i1], 1);
        tok_idx[i1 * kT + slot1] = token;
        tok_w  [i1 * kT + slot1] = w1;
    }
}

// ---------------------------------------------------------------------------
// Kernel 4: grouped expert GEMM. Block = (expert e, 64-token tile).
// 256 threads = 8 waves. x-tile staged into LDS with async global->LDS loads;
// SwiGLU intermediate h (64 x Fchunk) lives in LDS one F-chunk at a time;
// down-projection accumulates in registers across chunks (32 x v8f per lane),
// then routing-weighted global_atomic_add_f32 into out.
// Each weight B-fragment feeds FOUR WMMAs (row-blocks 0..3).
// ---------------------------------------------------------------------------
__global__ __launch_bounds__(256)
void moe_expert_kernel(const bf16* __restrict__ xb,
                       const bf16* __restrict__ wgT,   // [E][F][C]
                       const bf16* __restrict__ wuT,   // [E][F][C]
                       const bf16* __restrict__ wdT,   // [E][C][F]
                       const int*  __restrict__ counts,
                       const int*  __restrict__ tok_idx,
                       const float* __restrict__ tok_w,
                       float* __restrict__ out) {
    constexpr int XS_STRIDE = kC + 8;       // 1032 (keeps 16B alignment, skews banks)
    constexpr int HS_STRIDE = kFChunk + 8;  // 712
    __shared__ alignas(16) bf16 xs[kMTile * XS_STRIDE];   // ~132 KB
    __shared__ alignas(16) bf16 hs[kMTile * HS_STRIDE];   // ~91 KB
    __shared__ int   tidx_s[kMTile];
    __shared__ float tw_s[kMTile];

    const int e    = blockIdx.x >> 7;        // / 128
    const int tile = blockIdx.x & 127;
    const int cnt  = counts[e];
    if (tile * kMTile >= cnt) return;        // block-uniform

    const int tid  = threadIdx.x;
    const int lane = tid & 31;
    const int wv   = tid >> 5;

    if (tid < kMTile) {
        const int gi = tile * kMTile + tid;
        if (gi < cnt) {
            tidx_s[tid] = tok_idx[e * kT + gi];
            tw_s[tid]   = tok_w [e * kT + gi];
        } else {  // pad row: replicate last valid token, weight 0
            tidx_s[tid] = tok_idx[e * kT + cnt - 1];
            tw_s[tid]   = 0.0f;
        }
    }
    __syncthreads();

    // Gather 64 token rows of x (bf16) straight into LDS via async B128 copies
    // (GLOBAL_LOAD_ASYNC_TO_LDS_B128, tracked by ASYNCcnt). 4 threads/row.
    {
        const int row = tid >> 2;            // 0..63
        const int cb  = (tid & 3) * 8;       // 0..24
        const size_t src = (size_t)tidx_s[row] * kC;
#pragma unroll
        for (int j = 0; j < 32; ++j) {
            const int col = cb + j * 32;
            const unsigned lds_off =
                (unsigned)(uintptr_t)&xs[row * XS_STRIDE + col];  // addr[31:0] = LDS offset
            async_copy_b128(lds_off, &xb[src + col]);
        }
        wait_asynccnt0();
    }
    __syncthreads();

    const bf16* wgTe = wgT + (size_t)e * kF * kC;
    const bf16* wuTe = wuT + (size_t)e * kF * kC;
    const bf16* wdTe = wdT + (size_t)e * kC * kF;

    // Down-projection accumulators: wave wv owns output columns [wv*128, wv*128+128)
    // acc[4j+h] = row-block h (rows 16h..16h+15) of N-tile j.
    v8f acc[8 * kMBlk];
#pragma unroll
    for (int j = 0; j < 8 * kMBlk; ++j)
#pragma unroll
        for (int r = 0; r < 8; ++r) acc[j][r] = 0.0f;

#pragma unroll 1
    for (int ch = 0; ch < 4; ++ch) {
        const int fbase = ch * kFChunk;
        const int kc = (kF - fbase < kFChunk) ? (kF - fbase) : kFChunk; // 704/640
        const int nt = kc >> 4;

        // ---- Phase A: h[:, fbase:fbase+kc] = silu(x*Wg) * (x*Wu) into LDS ----
#pragma unroll 1
        for (int lt = wv; lt < nt; lt += 8) {
            const int f0 = fbase + lt * 16;
            // pull next tile's weights toward the caches while we compute
            if (lt + 8 < nt) {
                __builtin_prefetch(wgTe + (size_t)(f0 + 128) * kC, 0, 3);
                __builtin_prefetch(wuTe + (size_t)(f0 + 128) * kC, 0, 3);
            }
            v8f ag[kMBlk], au[kMBlk];
#pragma unroll
            for (int h = 0; h < kMBlk; ++h)
#pragma unroll
                for (int r = 0; r < 8; ++r) { ag[h][r] = 0.0f; au[h][r] = 0.0f; }

#pragma unroll 1
            for (int c0 = 0; c0 < kC; c0 += 32) {
                bf16x16 bg = load_frag(wgTe + (size_t)f0 * kC + c0, kC, lane);
                bf16x16 bu = load_frag(wuTe + (size_t)f0 * kC + c0, kC, lane);
#pragma unroll
                for (int h = 0; h < kMBlk; ++h) {
                    bf16x16 a = load_frag(xs + h * 16 * XS_STRIDE + c0, XS_STRIDE, lane);
                    ag[h] = wmma_bf16(a, bg, ag[h]);
                    au[h] = wmma_bf16(a, bu, au[h]);
                }
            }
            const int colb = lt * 16 + (lane & 15);
            const int rb   = (lane >> 4) * 8;
#pragma unroll
            for (int h = 0; h < kMBlk; ++h)
#pragma unroll
                for (int r = 0; r < 8; ++r) {
                    hs[(h * 16 + rb + r) * HS_STRIDE + colb] =
                        (bf16)fast_silu_mul(ag[h][r], au[h][r]);
                }
        }
        __syncthreads();

        // ---- Phase B: acc += h_chunk @ Wd[fbase:fbase+kc, :] ----
#pragma unroll 1
        for (int k0 = 0; k0 < kc; k0 += 32) {
            __builtin_prefetch(wdTe + (size_t)(wv * 128) * kF + fbase + k0 + 32, 0, 3);
            bf16x16 a[kMBlk];
#pragma unroll
            for (int h = 0; h < kMBlk; ++h)
                a[h] = load_frag(hs + h * 16 * HS_STRIDE + k0, HS_STRIDE, lane);
#pragma unroll
            for (int j = 0; j < 8; ++j) {
                const int n0 = (wv * 8 + j) * 16;
                bf16x16 b = load_frag(wdTe + (size_t)n0 * kF + fbase + k0, kF, lane);
#pragma unroll
                for (int h = 0; h < kMBlk; ++h)
                    acc[4 * j + h] = wmma_bf16(a[h], b, acc[4 * j + h]);
            }
        }
        __syncthreads();
    }

    // ---- Routing-weighted scatter-add into out ----
    const int rb = (lane >> 4) * 8;
    const int nn = lane & 15;
#pragma unroll
    for (int j = 0; j < 8; ++j) {
        const int n0 = (wv * 8 + j) * 16;
#pragma unroll
        for (int h = 0; h < kMBlk; ++h) {
#pragma unroll
            for (int r = 0; r < 8; ++r) {
                const int row = h * 16 + rb + r;
                atomicAdd(&out[(size_t)tidx_s[row] * kC + n0 + nn],
                          tw_s[row] * acc[4 * j + h][r]);
            }
        }
    }
}

// ---------------------------------------------------------------------------
// Host-side launch
// ---------------------------------------------------------------------------
extern "C" void kernel_launch(void* const* d_in, const int* in_sizes, int n_in,
                              void* d_out, int out_size, void* d_ws, size_t ws_size,
                              hipStream_t stream) {
    (void)in_sizes; (void)n_in; (void)out_size; (void)ws_size;

    const float* x      = (const float*)d_in[0];   // [T, C]
    const float* w_gate = (const float*)d_in[1];   // [E, C]
    const float* wg     = (const float*)d_in[2];   // [E, C, F]
    const float* wu     = (const float*)d_in[3];   // [E, C, F]
    const float* wd     = (const float*)d_in[4];   // [E, F, C]
    float* out          = (float*)d_out;           // [T, C]

    // Workspace carve-up (bf16 copies + routing tables)
    const size_t nXC  = (size_t)kT * kC;           // 8.4M
    const size_t nECF = (size_t)kE * kC * kF;      // 22.5M
    bf16* xb  = (bf16*)d_ws;
    bf16* wgT = xb  + nXC;
    bf16* wuT = wgT + nECF;
    bf16* wdT = wuT + nECF;
    int*  counts  = (int*)(wdT + nECF);
    int*  tok_idx = counts + 64;
    float* tok_w  = (float*)(tok_idx + (size_t)kE * kT);

    const dim3 blk(256);

    moe_zero_kernel<<<2048, blk, 0, stream>>>(out, counts, nXC);
    moe_cvt_bf16_kernel<<<2048, blk, 0, stream>>>(x, xb, nXC);
    moe_transpose_bf16_kernel<<<4096, blk, 0, stream>>>(wg, wgT, kC, kF); // -> [E][F][C]
    moe_transpose_bf16_kernel<<<4096, blk, 0, stream>>>(wu, wuT, kC, kF); // -> [E][F][C]
    moe_transpose_bf16_kernel<<<4096, blk, 0, stream>>>(wd, wdT, kF, kC); // -> [E][C][F]
    moe_gate_kernel<<<kT / 8, blk, 0, stream>>>(x, w_gate, counts, tok_idx, tok_w);
    moe_expert_kernel<<<kE * kMaxTiles, blk, 0, stream>>>(
        xb, wgT, wuT, wdT, counts, tok_idx, tok_w, out);
}
